// BridgeLayer4_xb_72000831750148
// MI455X (gfx1250) — compile-verified, matrix operand order checked by reading, not run on hardware
//
#include <hip/hip_runtime.h>
#include <hip/hip_bf16.h>

typedef __attribute__((ext_vector_type(16))) _Float16 v16h;
typedef __attribute__((ext_vector_type(8)))  _Float16 v8h;
typedef __attribute__((ext_vector_type(8)))  float    v8f;
typedef __attribute__((ext_vector_type(4)))  float    v4f;

static __device__ __forceinline__ float siluf(float x) { return x / (1.0f + __expf(-x)); }
static __device__ __forceinline__ float softplusf(float x) { return x > 20.0f ? x : log1pf(__expf(x)); }

// Async global->LDS copy of one DWORD per lane (gfx1250, tracked on ASYNCcnt).
// LDS aperture keeps the byte offset in the low 32 bits of a generic pointer.
static __device__ __forceinline__ void async_cp_b32(void* lds, const float* g)
{
    unsigned loff = (unsigned)(unsigned long long)lds;
    unsigned long long ga = (unsigned long long)g;
    asm volatile("global_load_async_to_lds_b32 %0, %1, off"
                 :: "v"(loff), "v"(ga) : "memory");
}
static __device__ __forceinline__ void async_wait0()
{
    asm volatile("s_wait_asynccnt 0" ::: "memory");
}

// ---------------------------------------------------------------------------
// Generic strided, batched WMMA GEMM:  C[m,n] = sum_k A[m,k] * B[k,n]
//   A elem: A + (z%aMod)*aStr + m*sAm + k*sAk
//   B elem: B + (z%bMod)*bStr + k*sBk + n*sBn
//   C elem: C + z*cStr        + m*sCm + n*sCn
// Interior tiles: per-lane async global->LDS (f32), s_wait_asynccnt + barrier,
// then fragments built from 4x ds_load_b128 + cvt per operand and fed to
// v_wmma_f32_16x16x32_f16 (f32 accumulate). Edge tiles: guarded scalar path.
// 128 threads = 4 waves; each wave owns a 16x16 tile of a 32x32 block tile.
// ---------------------------------------------------------------------------
__global__ void __launch_bounds__(128)
gemm_wmma_f32(const float* __restrict__ A, const float* __restrict__ B,
              float* __restrict__ C,
              int M, int N, int K,
              long sAm, long sAk, long sBk, long sBn, long sCm, long sCn,
              int aMod, long aStr, int bMod, long bStr, long cStr)
{
    __shared__ alignas(16) float Af[32][36];    // [m][k]  (rows 144B -> 16B aligned)
    __shared__ alignas(16) float Bf[32][36];    // [n][k]  (B stored transposed)

    const int tid  = threadIdx.x;
    const int lane = tid & 31;
    const int wave = tid >> 5;
    const int wm   = (wave >> 1) * 16;
    const int wn   = (wave & 1) * 16;

    const int tm = blockIdx.y * 32;
    const int tn = blockIdx.x * 32;
    const int z  = blockIdx.z;

    A += (long)(z % aMod) * aStr;
    B += (long)(z % bMod) * bStr;
    C += (long)z * cStr;

    const bool mfull = (tm + 32 <= M);
    const bool nfull = (tn + 32 <= N);
    const bool ak1 = (sAk == 1);   // adjacent lanes walk the unit-stride dim
    const bool bn1 = (sBn == 1);

    v8f acc = {};

    for (int kt = 0; kt < K; kt += 32) {
        const bool kfull = (kt + 32 <= K);
        if (mfull && nfull && kfull) {
            if (ak1) {
#pragma unroll
                for (int it = 0; it < 8; ++it) {
                    int idx = tid + it * 128;
                    int rw = idx >> 5, q = idx & 31;          // rw=m, q=k (lanes walk k)
                    async_cp_b32(&Af[rw][q], &A[(long)(tm + rw) * sAm + (long)(kt + q)]);
                }
            } else {
#pragma unroll
                for (int it = 0; it < 8; ++it) {
                    int idx = tid + it * 128;
                    int rw = idx & 31, q = idx >> 5;          // lanes walk m
                    async_cp_b32(&Af[rw][q], &A[(long)(tm + rw) * sAm + (long)(kt + q) * sAk]);
                }
            }
            if (bn1) {
#pragma unroll
                for (int it = 0; it < 8; ++it) {
                    int idx = tid + it * 128;
                    int rw = idx >> 5, q = idx & 31;          // rw=k, q=n (lanes walk n)
                    async_cp_b32(&Bf[q][rw], &B[(long)(kt + rw) * sBk + (long)(tn + q)]);
                }
            } else {
#pragma unroll
                for (int it = 0; it < 8; ++it) {
                    int idx = tid + it * 128;
                    int rw = idx & 31, q = idx >> 5;          // lanes walk k
                    async_cp_b32(&Bf[q][rw], &B[(long)(kt + rw) * sBk + (long)(tn + q) * sBn]);
                }
            }
        } else {
#pragma unroll
            for (int it = 0; it < 8; ++it) {
                int idx = tid + it * 128;
                int rw = idx >> 5, q = idx & 31;
                int m = tm + rw, k = kt + q;
                Af[rw][q] = (m < M && k < K) ? A[(long)m * sAm + (long)k * sAk] : 0.0f;
                int kk = kt + rw, n = tn + q;
                Bf[q][rw] = (kk < K && n < N) ? B[(long)kk * sBk + (long)n * sBn] : 0.0f;
            }
        }
        async_wait0();
        __syncthreads();

        // Fragment layout (16-bit A 16x32): lane m=lane&15; K runs
        // [kb0, kb0+8) in v0..3 and [kb0+16, kb0+24) in v4..7, kb0 = lane>=16 ? 8 : 0.
        const int kb0  = (lane >= 16) ? 8 : 0;
        const int arow = wm + (lane & 15);
        const int brow = wn + (lane & 15);
        v4f a00 = *(const v4f*)&Af[arow][kb0];
        v4f a01 = *(const v4f*)&Af[arow][kb0 + 4];
        v4f a10 = *(const v4f*)&Af[arow][kb0 + 16];
        v4f a11 = *(const v4f*)&Af[arow][kb0 + 20];
        v4f b00 = *(const v4f*)&Bf[brow][kb0];
        v4f b01 = *(const v4f*)&Bf[brow][kb0 + 4];
        v4f b10 = *(const v4f*)&Bf[brow][kb0 + 16];
        v4f b11 = *(const v4f*)&Bf[brow][kb0 + 20];
        v16h af, bf;
#pragma unroll
        for (int e = 0; e < 4; ++e) {
            af[e]      = (_Float16)a00[e];
            af[4 + e]  = (_Float16)a01[e];
            af[8 + e]  = (_Float16)a10[e];
            af[12 + e] = (_Float16)a11[e];
            bf[e]      = (_Float16)b00[e];
            bf[4 + e]  = (_Float16)b01[e];
            bf[8 + e]  = (_Float16)b10[e];
            bf[12 + e] = (_Float16)b11[e];
        }
        acc = __builtin_amdgcn_wmma_f32_16x16x32_f16(false, af, false, bf,
                                                     (short)0, acc, false, false);
        __syncthreads();
    }

    if (mfull && nfull) {
#pragma unroll
        for (int rr = 0; rr < 8; ++rr) {
            int m = tm + wm + rr + ((lane >= 16) ? 8 : 0);
            int n = tn + wn + (lane & 15);
            C[(long)m * sCm + (long)n * sCn] = acc[rr];
        }
    } else {
#pragma unroll
        for (int rr = 0; rr < 8; ++rr) {
            int m = tm + wm + rr + ((lane >= 16) ? 8 : 0);
            int n = tn + wn + (lane & 15);
            if (m < M && n < N) C[(long)m * sCm + (long)n * sCn] = acc[rr];
        }
    }
}

// ---------------------------------------------------------------------------
// Depthwise convolutions + bias + SiLU. xz is channel-major [b][2d][L],
// channels [0,d) are the conv branch; output xc is [b][d][L].
// ---------------------------------------------------------------------------
__global__ void dwconv2d_silu(const float* __restrict__ xz, const float* __restrict__ wconv,
                              const float* __restrict__ bconv, float* __restrict__ xc,
                              int Bb, int d, int H, int W)
{
    long L = (long)H * W;
    long tot = (long)Bb * d * L;
    long i = blockIdx.x * (long)blockDim.x + threadIdx.x;
    if (i >= tot) return;
    int  l = (int)(i % L);
    long t = i / L;
    int  ch = (int)(t % d);
    int  b  = (int)(t / d);
    int  h = l / W, w = l % W;
    const float* xin = xz + ((long)b * (2 * d) + ch) * L;
    float s = bconv[ch];
#pragma unroll
    for (int kh = 0; kh < 3; ++kh) {
        int hh = h + kh - 1;
        if (hh < 0 || hh >= H) continue;
#pragma unroll
        for (int kw = 0; kw < 3; ++kw) {
            int ww = w + kw - 1;
            if (ww < 0 || ww >= W) continue;
            s += wconv[(kh * 3 + kw) * d + ch] * xin[(long)hh * W + ww];
        }
    }
    xc[i] = siluf(s);
}

__global__ void dwconv1d_silu(const float* __restrict__ xz, const float* __restrict__ wconv,
                              const float* __restrict__ bconv, float* __restrict__ xc,
                              int Bb, int d, int L)
{
    long tot = (long)Bb * d * L;
    long i = blockIdx.x * (long)blockDim.x + threadIdx.x;
    if (i >= tot) return;
    int  l = (int)(i % L);
    long t = i / L;
    int  ch = (int)(t % d);
    int  b  = (int)(t / d);
    const float* xin = xz + ((long)b * (2 * d) + ch) * L;
    float s = bconv[ch];
#pragma unroll
    for (int kw = 0; kw < 3; ++kw) {
        int ll = l + kw - 1;
        if (ll < 0 || ll >= L) continue;
        s += wconv[kw * d + ch] * xin[ll];
    }
    xc[i] = siluf(s);
}

// ---------------------------------------------------------------------------
// Build direction-rearranged xs[b][k][d][L] from xc[b][d][L].
// kdirs==4: {hw, wh, rev(hw), rev(wh)}   kdirs==2: {fwd, rev}
// ---------------------------------------------------------------------------
__global__ void build_xs(const float* __restrict__ xc, float* __restrict__ xs,
                         int Bb, int kdirs, int d, int H, int W)
{
    long L = (long)H * W;
    long tot = (long)Bb * kdirs * d * L;
    long i = blockIdx.x * (long)blockDim.x + threadIdx.x;
    if (i >= tot) return;
    long l = i % L;
    long t = i / L;
    int  dd = (int)(t % d); t /= d;
    int  k  = (int)(t % kdirs);
    int  b  = (int)(t / kdirs);
    long lm;
    if (kdirs == 4) {
        if (k == 0) lm = l;
        else if (k == 1) { long hh = l % H, ww = l / H; lm = hh * W + ww; }
        else if (k == 2) lm = L - 1 - l;
        else { long l2 = L - 1 - l; long hh = l2 % H, ww = l2 / H; lm = hh * W + ww; }
    } else {
        lm = (k == 0) ? l : (L - 1 - l);
    }
    xs[i] = xc[((long)b * d + dd) * L + lm];
}

// ---------------------------------------------------------------------------
// delta = softplus(dt_w @ dts + dt_b).
// x_dbl layout: [b][k][L][c]  (c-contiguous rows: dts/B/C contiguous per step)
// ---------------------------------------------------------------------------
__global__ void dt_softplus(const float* __restrict__ xdbl, const float* __restrict__ dtw,
                            const float* __restrict__ dtb, float* __restrict__ delta,
                            int Bb, int kdirs, int d, int r, int c, int L)
{
    long tot = (long)Bb * kdirs * d * L;
    long i = blockIdx.x * (long)blockDim.x + threadIdx.x;
    if (i >= tot) return;
    long l = i % L;
    long t = i / L;
    int  dd = (int)(t % d); t /= d;
    int  kk = (int)(t % kdirs);
    int  b  = (int)(t / kdirs);
    const float* xp = xdbl + (((long)b * kdirs + kk) * L + l) * c;
    const float* wp = dtw + ((long)kk * d + dd) * r;
    float s = dtb[kk * d + dd];
    for (int rr = 0; rr < r; ++rr) s += wp[rr] * xp[rr];
    delta[i] = softplusf(s);
}

// ---------------------------------------------------------------------------
// Chunked selective scan (h' = exp(dt*A)*h + dt*u*B is a diagonal linear
// recurrence => associative). Pass1: per-chunk (prodA, accB). Seq: compose
// chunk-initial states. Pass2: replay chunks emitting y = <h,C> + D*u.
// x_dbl is [b][k][L][c]; B_t = row+r (16 contiguous), C_t = row+r+16.
// ---------------------------------------------------------------------------
__global__ void scan_pass1(const float* __restrict__ xs, const float* __restrict__ delta,
                           const float* __restrict__ xdbl, const float* __restrict__ Alog,
                           float* __restrict__ cp, float* __restrict__ ca,
                           int Bb, int kdirs, int d, int r, int c, int L, int NC, int CL)
{
    int tot = Bb * kdirs * d * NC;
    int i = blockIdx.x * blockDim.x + threadIdx.x;
    if (i >= tot) return;
    int ch = i % NC; int t = i / NC;
    int dd = t % d;  t /= d;
    int kk = t % kdirs;
    int b  = t / kdirs;
    float Aa[16], P[16], Bc[16];
#pragma unroll
    for (int nn = 0; nn < 16; ++nn) {
        Aa[nn] = -__expf(Alog[((long)kk * d + dd) * 16 + nn]);
        P[nn] = 1.0f; Bc[nn] = 0.0f;
    }
    long base = (((long)b * kdirs + kk) * d + dd) * L;
    const float* u  = xs + base;
    const float* dt = delta + base;
    const float* xrow = xdbl + ((long)b * kdirs + kk) * L * (long)c + r;
    int l0 = ch * CL, l1 = l0 + CL; if (l1 > L) l1 = L;
    for (int l = l0; l < l1; ++l) {
        float dtv = dt[l];
        float du  = dtv * u[l];
        const float* Bt = xrow + (long)l * c;
#pragma unroll
        for (int nn = 0; nn < 16; ++nn) {
            float a = __expf(dtv * Aa[nn]);
            P[nn] *= a;
            Bc[nn] = Bc[nn] * a + du * Bt[nn];
        }
    }
    float* cpp = cp + (long)i * 16;
    float* cap = ca + (long)i * 16;
#pragma unroll
    for (int nn = 0; nn < 16; ++nn) { cpp[nn] = P[nn]; cap[nn] = Bc[nn]; }
}

__global__ void scan_seq(const float* __restrict__ cp, const float* __restrict__ ca,
                         float* __restrict__ h0, int tot, int NC)
{
    int i = blockIdx.x * blockDim.x + threadIdx.x;
    if (i >= tot) return;
    float h[16];
#pragma unroll
    for (int nn = 0; nn < 16; ++nn) h[nn] = 0.0f;
    long base = (long)i * NC * 16;
    for (int ch = 0; ch < NC; ++ch) {
        long o = base + (long)ch * 16;
#pragma unroll
        for (int nn = 0; nn < 16; ++nn) h0[o + nn] = h[nn];
#pragma unroll
        for (int nn = 0; nn < 16; ++nn) h[nn] = cp[o + nn] * h[nn] + ca[o + nn];
    }
}

__global__ void scan_pass2(const float* __restrict__ xs, const float* __restrict__ delta,
                           const float* __restrict__ xdbl, const float* __restrict__ Alog,
                           const float* __restrict__ Dp, const float* __restrict__ h0,
                           float* __restrict__ y,
                           int Bb, int kdirs, int d, int r, int c, int L, int NC, int CL)
{
    int tot = Bb * kdirs * d * NC;
    int i = blockIdx.x * blockDim.x + threadIdx.x;
    if (i >= tot) return;
    int ch = i % NC; int t = i / NC;
    int dd = t % d;  t /= d;
    int kk = t % kdirs;
    int b  = t / kdirs;
    float Aa[16], h[16];
#pragma unroll
    for (int nn = 0; nn < 16; ++nn) {
        Aa[nn] = -__expf(Alog[((long)kk * d + dd) * 16 + nn]);
        h[nn]  = h0[(long)i * 16 + nn];
    }
    float Dv = Dp[kk * d + dd];
    long base = (((long)b * kdirs + kk) * d + dd) * L;
    const float* u  = xs + base;
    const float* dt = delta + base;
    const float* xrow = xdbl + ((long)b * kdirs + kk) * L * (long)c + r;
    float* yo = y + base;
    int l0 = ch * CL, l1 = l0 + CL; if (l1 > L) l1 = L;
    for (int l = l0; l < l1; ++l) {
        float dtv = dt[l];
        float uv  = u[l];
        float du  = dtv * uv;
        const float* Bt = xrow + (long)l * c;
        const float* Ct = Bt + 16;
        float acc = 0.0f;
#pragma unroll
        for (int nn = 0; nn < 16; ++nn) {
            h[nn] = h[nn] * __expf(dtv * Aa[nn]) + du * Bt[nn];
            acc += h[nn] * Ct[nn];
        }
        yo[l] = acc + Dv * uv;
    }
}

// ---------------------------------------------------------------------------
// Direction combine. 2D: y0[l] + y2[L-1-l] + y1[w*H+h] + y3[L-1-(w*H+h)].
// 1D: y0[l] + y1[L-1-l]. Output yc is ROW-major (b, l, d).
// ---------------------------------------------------------------------------
__global__ void combine2d(const float* __restrict__ y, float* __restrict__ yc,
                          int Bb, int d, int H, int W)
{
    long L = (long)H * W;
    long tot = (long)Bb * d * L;
    long i = blockIdx.x * (long)blockDim.x + threadIdx.x;
    if (i >= tot) return;
    long l = i % L;
    long t = i / L;
    int  dd = (int)(t % d);
    int  b  = (int)(t / d);
    long h = l / W, w = l % W;
    long lw = w * H + h;
    const float* yb = y + (long)b * 4 * d * L;
    float v = yb[(long)dd * L + l]
            + yb[((long)2 * d + dd) * L + (L - 1 - l)]
            + yb[((long)1 * d + dd) * L + lw]
            + yb[((long)3 * d + dd) * L + (L - 1 - lw)];
    yc[((long)b * L + l) * d + dd] = v;
}

__global__ void combine1d(const float* __restrict__ y, float* __restrict__ yc,
                          int Bb, int d, int L)
{
    long tot = (long)Bb * d * L;
    long i = blockIdx.x * (long)blockDim.x + threadIdx.x;
    if (i >= tot) return;
    long l = i % L;
    long t = i / L;
    int  dd = (int)(t % d);
    int  b  = (int)(t / d);
    const float* yb = y + (long)b * 2 * d * L;
    float v = yb[(long)dd * L + l] + yb[((long)d + dd) * L + (L - 1 - l)];
    yc[((long)b * L + l) * d + dd] = v;
}

// ---------------------------------------------------------------------------
// Fused layernorm (over last dim of row-major yc) * silu(z), z strided from xz.
// ---------------------------------------------------------------------------
__global__ void __launch_bounds__(128)
ln_gate(const float* __restrict__ yc, const float* __restrict__ xz,
        const float* __restrict__ g, const float* __restrict__ bt,
        float* __restrict__ out, int d, int L, float eps)
{
    long row = blockIdx.x;
    int  b = (int)(row / L);
    long l = row % L;
    const float* x = yc + row * d;
    __shared__ float red[128];
    float s = 0.0f;
    for (int j = threadIdx.x; j < d; j += 128) s += x[j];
    red[threadIdx.x] = s; __syncthreads();
    for (int st = 64; st > 0; st >>= 1) {
        if (threadIdx.x < st) red[threadIdx.x] += red[threadIdx.x + st];
        __syncthreads();
    }
    float mean = red[0] / d;
    __syncthreads();
    float v = 0.0f;
    for (int j = threadIdx.x; j < d; j += 128) { float dx = x[j] - mean; v += dx * dx; }
    red[threadIdx.x] = v; __syncthreads();
    for (int st = 64; st > 0; st >>= 1) {
        if (threadIdx.x < st) red[threadIdx.x] += red[threadIdx.x + st];
        __syncthreads();
    }
    float inv = rsqrtf(red[0] / d + eps);
    const float* zrow = xz + ((long)b * (2 * d) + d) * L + l;
    for (int j = threadIdx.x; j < d; j += 128) {
        float zn = zrow[(long)j * L];
        out[row * d + j] = ((x[j] - mean) * inv * g[j] + bt[j]) * siluf(zn);
    }
}

// Plain layernorm over row-major rows of width Dw (for F before the 1D pass).
__global__ void __launch_bounds__(64)
ln_rows(const float* __restrict__ x, const float* __restrict__ g,
        const float* __restrict__ bt, float* __restrict__ out, int Dw, float eps)
{
    long row = blockIdx.x;
    const float* xr = x + row * Dw;
    __shared__ float red[64];
    float s = 0.0f;
    for (int j = threadIdx.x; j < Dw; j += 64) s += xr[j];
    red[threadIdx.x] = s; __syncthreads();
    for (int st = 32; st > 0; st >>= 1) {
        if (threadIdx.x < st) red[threadIdx.x] += red[threadIdx.x + st];
        __syncthreads();
    }
    float mean = red[0] / Dw;
    __syncthreads();
    float v = 0.0f;
    for (int j = threadIdx.x; j < Dw; j += 64) { float dx = xr[j] - mean; v += dx * dx; }
    red[threadIdx.x] = v; __syncthreads();
    for (int st = 32; st > 0; st >>= 1) {
        if (threadIdx.x < st) red[threadIdx.x] += red[threadIdx.x + st];
        __syncthreads();
    }
    float inv = rsqrtf(red[0] / Dw + eps);
    for (int j = threadIdx.x; j < Dw; j += 64)
        out[row * Dw + j] = (xr[j] - mean) * inv * g[j] + bt[j];
}

// Final residual add + flat region split (F layout makes reshapes flat copies).
__global__ void add_region(float* __restrict__ dst, const float* __restrict__ F,
                           const float* __restrict__ Dl, long off, long RB, long FB, int Bb)
{
    long tot = (long)Bb * RB;
    long i = blockIdx.x * (long)blockDim.x + threadIdx.x;
    if (i >= tot) return;
    long b = i / RB, j = i % RB;
    long src = b * FB + off + j;
    dst[i] = F[src] + Dl[src];
}

// ---------------------------------------------------------------------------
// Host-side orchestration
// ---------------------------------------------------------------------------
struct SSP {
    const float *w_in, *w_conv, *b_conv, *x_proj_w, *dt_w, *dt_b, *A_log, *D, *ln_g, *ln_b, *w_out;
};

static SSP getp(void* const* d_in, int base) {
    SSP p;
    p.w_in     = (const float*)d_in[base + 0];
    p.w_conv   = (const float*)d_in[base + 1];
    p.b_conv   = (const float*)d_in[base + 2];
    p.x_proj_w = (const float*)d_in[base + 3];
    p.dt_w     = (const float*)d_in[base + 4];
    p.dt_b     = (const float*)d_in[base + 5];
    p.A_log    = (const float*)d_in[base + 6];
    p.D        = (const float*)d_in[base + 7];
    p.ln_g     = (const float*)d_in[base + 8];
    p.ln_b     = (const float*)d_in[base + 9];
    p.w_out    = (const float*)d_in[base + 10];
    return p;
}

static inline long cdivl(long a, long b) { return (a + b - 1) / b; }

static void launch_gemm(hipStream_t s, const float* A, const float* B, float* C,
                        int M, int N, int K,
                        long sAm, long sAk, long sBk, long sBn, long sCm, long sCn,
                        int aMod, long aStr, int bMod, long bStr, long cStr, int batches)
{
    dim3 g((unsigned)cdivl(N, 32), (unsigned)cdivl(M, 32), (unsigned)batches);
    gemm_wmma_f32<<<g, 128, 0, s>>>(A, B, C, M, N, K, sAm, sAk, sBk, sBn, sCm, sCn,
                                    aMod, aStr, bMod, bStr, cStr);
}

static void run_ss2d(hipStream_t s, const float* x, int conv2d, int kdirs,
                     const SSP& p, int Bb, int dm, int H, int W,
                     long inAm, long inAk, long inAstr,
                     float* Cout, long outCm, long outCn, long outCstr,
                     float* scratch)
{
    const int d = 2 * dm;
    int r = (dm + 15) / 16; if (r < 1) r = 1;
    const int c = r + 32;
    const long L = (long)H * W;
    const int NC = 64;
    const int CL = (int)cdivl(L, NC);

    long o = 0;
    float* xz    = scratch + o; o += (long)Bb * 2 * d * L;
    float* xc    = scratch + o; o += (long)Bb * d * L;
    float* xs    = scratch + o; o += (long)Bb * kdirs * d * L;
    float* xdbl  = scratch + o; o += (long)Bb * kdirs * c * L;
    float* delta = scratch + o; o += (long)Bb * kdirs * d * L;
    float* ysc   = scratch + o; o += (long)Bb * kdirs * d * L;
    long nst = (long)Bb * kdirs * d * NC * 16;
    float* cp = scratch + o; o += nst;
    float* ca = scratch + o; o += nst;
    float* h0 = scratch + o; o += nst;
    float* ycomb = xc;  // reuse: xc dead after build_xs
    float* tg    = xs;  // reuse: xs dead after scan

    const int BIG = 1 << 30;

    // 1) xz[b][n][l] = sum_k x(b,l,k) * w_in[n,k]
    launch_gemm(s, x, p.w_in, xz, (int)L, 2 * d, dm,
                inAm, inAk, 1, dm, 1, L,
                BIG, inAstr, 1, 0, (long)2 * d * L, Bb);

    // 2) depthwise conv + SiLU
    {
        long tot = (long)Bb * d * L;
        dim3 g((unsigned)cdivl(tot, 256));
        if (conv2d) dwconv2d_silu<<<g, 256, 0, s>>>(xz, p.w_conv, p.b_conv, xc, Bb, d, H, W);
        else        dwconv1d_silu<<<g, 256, 0, s>>>(xz, p.w_conv, p.b_conv, xc, Bb, d, (int)L);
    }

    // 3) direction rearrangement
    {
        long tot = (long)Bb * kdirs * d * L;
        build_xs<<<(unsigned)cdivl(tot, 256), 256, 0, s>>>(xc, xs, Bb, kdirs, d, H, W);
    }

    // 4) x_dbl[b][k][L][c] = (x_proj_w[k] @ xs[b][k])^T  (c-contiguous rows)
    launch_gemm(s, p.x_proj_w, xs, xdbl, c, (int)L, d,
                d, 1, L, 1, 1, c,
                kdirs, (long)c * d, BIG, (long)d * L, (long)c * L, Bb * kdirs);

    // 5) delta = softplus(dt_w @ dts + dt_b)
    {
        long tot = (long)Bb * kdirs * d * L;
        dt_softplus<<<(unsigned)cdivl(tot, 256), 256, 0, s>>>(xdbl, p.dt_w, p.dt_b, delta,
                                                              Bb, kdirs, d, r, c, (int)L);
    }

    // 6) chunk-parallel selective scan
    {
        int tot1 = Bb * kdirs * d * NC;
        scan_pass1<<<(unsigned)cdivl(tot1, 128), 128, 0, s>>>(xs, delta, xdbl, p.A_log, cp, ca,
                                                              Bb, kdirs, d, r, c, (int)L, NC, CL);
        int tot0 = Bb * kdirs * d;
        scan_seq<<<(unsigned)cdivl(tot0, 128), 128, 0, s>>>(cp, ca, h0, tot0, NC);
        scan_pass2<<<(unsigned)cdivl(tot1, 128), 128, 0, s>>>(xs, delta, xdbl, p.A_log, p.D, h0,
                                                              ysc, Bb, kdirs, d, r, c, (int)L, NC, CL);
    }

    // 7) combine directions -> row-major (b, l, d)
    {
        long tot = (long)Bb * d * L;
        if (kdirs == 4) combine2d<<<(unsigned)cdivl(tot, 256), 256, 0, s>>>(ysc, ycomb, Bb, d, H, W);
        else            combine1d<<<(unsigned)cdivl(tot, 256), 256, 0, s>>>(ysc, ycomb, Bb, d, (int)L);
    }

    // 8) layernorm + silu(z) gate
    ln_gate<<<(unsigned)((long)Bb * L), 128, 0, s>>>(ycomb, xz, p.ln_g, p.ln_b, tg, d, (int)L, 1e-5f);

    // 9) out = tg @ w_out^T
    launch_gemm(s, tg, p.w_out, Cout, (int)L, dm, d,
                d, 1, 1, d, outCm, outCn,
                BIG, (long)L * d, 1, 0, outCstr, Bb);
}

extern "C" void kernel_launch(void* const* d_in, const int* in_sizes, int n_in,
                              void* d_out, int out_size, void* d_ws, size_t ws_size,
                              hipStream_t stream)
{
    (void)in_sizes; (void)n_in; (void)out_size;

    const float* C1 = (const float*)d_in[0];
    const float* C2 = (const float*)d_in[1];
    const float* C3 = (const float*)d_in[2];
    const float* C4 = (const float*)d_in[3];
    SSP p1 = getp(d_in, 4);
    SSP p2 = getp(d_in, 15);
    SSP p3 = getp(d_in, 26);
    SSP p4 = getp(d_in, 37);
    SSP pf = getp(d_in, 48);
    const float* g1 = (const float*)d_in[59];
    const float* b1 = (const float*)d_in[60];

    const int Bb = 2, dims = 24, H = 112, W = 112;
    const long HW = (long)H * W;
    const long Ltot = HW + HW / 2 + HW / 4 + HW / 8;        // 23520
    const long FB = Ltot * dims;                             // per-batch F size

    long rbs[4]  = { dims * HW, (2 * dims) * (HW / 4), (4 * dims) * (HW / 16), (8 * dims) * (HW / 64) };
    long offs[4] = { 0, rbs[0], rbs[0] + rbs[1], rbs[0] + rbs[1] + rbs[2] };

    float* F   = (float*)d_ws;
    float* Fln = F   + (long)Bb * FB;
    float* D1  = Fln + (long)Bb * FB;
    float* scratch = D1 + (long)Bb * FB;

    size_t need = ((size_t)(3L * Bb * FB) + 26000000uL) * sizeof(float);
    if (ws_size < need) return;

    run_ss2d(stream, C1, 1, 4, p1, Bb, dims,     H,     W,
             1, HW,        (long)dims * HW,
             F + offs[0], 1, HW,        FB, scratch);
    run_ss2d(stream, C2, 1, 4, p2, Bb, 2 * dims, H / 2, W / 2,
             1, HW / 4,    (long)2 * dims * (HW / 4),
             F + offs[1], 1, HW / 4,    FB, scratch);
    run_ss2d(stream, C3, 1, 4, p3, Bb, 4 * dims, H / 4, W / 4,
             1, HW / 16,   (long)4 * dims * (HW / 16),
             F + offs[2], 1, HW / 16,   FB, scratch);
    run_ss2d(stream, C4, 1, 4, p4, Bb, 8 * dims, H / 8, W / 8,
             1, HW / 64,   (long)8 * dims * (HW / 64),
             F + offs[3], 1, HW / 64,   FB, scratch);

    ln_rows<<<(unsigned)((long)Bb * Ltot), 64, 0, stream>>>(F, g1, b1, Fln, dims, 1e-5f);
    run_ss2d(stream, Fln, 0, 2, pf, Bb, dims, 1, (int)Ltot,
             dims, 1, Ltot * dims,
             D1, dims, 1, FB, scratch);

    long dst_off = 0;
    for (int i = 0; i < 4; ++i) {
        long tot = (long)Bb * rbs[i];
        add_region<<<(unsigned)cdivl(tot, 256), 256, 0, stream>>>(
            (float*)d_out + dst_off, F, D1, offs[i], rbs[i], FB, Bb);
        dst_off += tot;
    }
}
